// MultiHeadAttention_63771674411316
// MI455X (gfx1250) — compile-verified, matrix operand order checked by reading, not run on hardware
//
#include <hip/hip_runtime.h>

// ---------------------------------------------------------------------------
// MI455X (gfx1250) multi-head attention, bf16 WMMA pipeline, round 3.
//   B=2, S=4096, E=768, H=12, dk=64.  ~142 GFLOP, compute-bound.
//   Changes vs round 2:
//     * Flash-attention K/V tile staging now uses the CDNA5 async copy path:
//       global_load_async_to_lds_b128 (ASYNCcnt) instead of
//       global->VGPR->ds_store.  Frees staging registers, removes ds_store
//       traffic, transfers overlap the whole compute of the previous tile,
//       synchronized with s_wait_asynccnt + workgroup barrier.
// ---------------------------------------------------------------------------

typedef __attribute__((ext_vector_type(16))) __bf16        v16bf;
typedef __attribute__((ext_vector_type(8)))  float         v8f;
typedef __attribute__((ext_vector_type(4)))  unsigned int  u32x4;
typedef __attribute__((ext_vector_type(8)))  unsigned int  u32x8;

#define D_EMB 768
#define SEQ   4096
#define NH    12
#define DK    64

__device__ __forceinline__ unsigned short f2bf(float x) {
    unsigned int u = __builtin_bit_cast(unsigned int, x);
    u += 0x7FFFu + ((u >> 16) & 1u);        // round-to-nearest-even
    return (unsigned short)(u >> 16);
}

// Load one 16x32 bf16 A-fragment (or mirrored B-fragment) from a row-major
// matrix of bf16 (stored as u16), leading dimension `ld` (in elements).
// ISA layout: lanes 0-15 hold row (row0+lane), K chunks {0..7,16..23};
//             lanes 16-31 hold row (row0+lane-16), K chunks {8..15,24..31}.
__device__ __forceinline__ v16bf load_frag(const unsigned short* mat,
                                           int row0, int ld, int kbase) {
    const int lane = threadIdx.x & 31;
    const unsigned short* p =
        mat + (size_t)(row0 + (lane & 15)) * ld + kbase + ((lane >> 4) << 3);
    u32x4 lo = *(const u32x4*)p;          // 8 halves: k .. k+7
    u32x4 hi = *(const u32x4*)(p + 16);   // 8 halves: k+16 .. k+23
    u32x8 w;
    w[0] = lo[0]; w[1] = lo[1]; w[2] = lo[2]; w[3] = lo[3];
    w[4] = hi[0]; w[5] = hi[1]; w[6] = hi[2]; w[7] = hi[3];
    return __builtin_bit_cast(v16bf, w);
}

__device__ __forceinline__ v8f wmma_bf16(v16bf a, v16bf b, v8f c) {
    return __builtin_amdgcn_wmma_f32_16x16x32_bf16(
        /*neg_a=*/false, a, /*neg_b=*/false, b,
        /*c_mod=*/(short)0, c, /*reuse_a=*/false, /*reuse_b=*/false);
}

// Async 16-byte copy: global -> LDS, per lane, tracked by ASYNCcnt.
// VDST = 32-bit LDS byte address VGPR, VADDR = 64-bit global address (GV mode).
__device__ __forceinline__ void async_ld16(unsigned short* ldsp,
                                           const unsigned short* gptr) {
    unsigned int       laddr = (unsigned int)(size_t)ldsp;       // addr[31:0]
    unsigned long long gaddr = (unsigned long long)(size_t)gptr;
    asm volatile("global_load_async_to_lds_b128 %0, %1, off"
                 :: "v"(laddr), "v"(gaddr) : "memory");
}

__device__ __forceinline__ void wait_async0() {
    asm volatile("s_wait_asynccnt 0x0" ::: "memory");
}

// ---------------------------------------------------------------------------
// Pack kernels
// ---------------------------------------------------------------------------
__global__ void mha_cvt_f32_bf16(const float* __restrict__ x,
                                 unsigned short* __restrict__ y, int n) {
    int i = blockIdx.x * 256 + threadIdx.x;
    if (i < n) y[i] = f2bf(x[i]);
}

__global__ void mha_transpose_w(const float* __restrict__ w,
                                unsigned short* __restrict__ wt) {
    int i = blockIdx.x * 256 + threadIdx.x;      // i < 768*768
    int k = i / D_EMB, n = i - k * D_EMB;
    wt[(size_t)n * D_EMB + k] = f2bf(w[i]);
}

// ---------------------------------------------------------------------------
// QKV projection, software-pipelined. Wave computes a 16x64 tile.
// Q pre-scaled by 1/sqrt(dk)=0.125.  Q,K -> [B,H,S,64]; V -> [B,H,64,S].
// ---------------------------------------------------------------------------
__global__ void __launch_bounds__(128)
mha_qkv_gemm(const unsigned short* __restrict__ Xb,
             const unsigned short* __restrict__ Wt,
             unsigned short* __restrict__ Qp,
             unsigned short* __restrict__ Kp,
             unsigned short* __restrict__ Vtp) {
    const int wave = threadIdx.x >> 5;
    const int lane = threadIdx.x & 31;
    const int m0 = blockIdx.x * 64 + wave * 16;
    const int n0 = blockIdx.y * 64;
    const int z  = blockIdx.z;
    const unsigned short* W = Wt + (size_t)z * D_EMB * D_EMB;

    v8f c0 = {}, c1 = {}, c2 = {}, c3 = {};
    v16bf a  = load_frag(Xb, m0, D_EMB, 0);
    v16bf b0 = load_frag(W, n0,      D_EMB, 0);
    v16bf b1 = load_frag(W, n0 + 16, D_EMB, 0);
    v16bf b2 = load_frag(W, n0 + 32, D_EMB, 0);
    v16bf b3 = load_frag(W, n0 + 48, D_EMB, 0);
    for (int k = 32; k < D_EMB; k += 32) {
        v16bf an  = load_frag(Xb, m0, D_EMB, k);      // prefetch next step
        v16bf bn0 = load_frag(W, n0,      D_EMB, k);
        v16bf bn1 = load_frag(W, n0 + 16, D_EMB, k);
        v16bf bn2 = load_frag(W, n0 + 32, D_EMB, k);
        v16bf bn3 = load_frag(W, n0 + 48, D_EMB, k);
        c0 = wmma_bf16(a, b0, c0);
        c1 = wmma_bf16(a, b1, c1);
        c2 = wmma_bf16(a, b2, c2);
        c3 = wmma_bf16(a, b3, c3);
        a = an; b0 = bn0; b1 = bn1; b2 = bn2; b3 = bn3;
    }
    c0 = wmma_bf16(a, b0, c0);
    c1 = wmma_bf16(a, b1, c1);
    c2 = wmma_bf16(a, b2, c2);
    c3 = wmma_bf16(a, b3, c3);

    const int hi = (lane >> 4) << 3;
    const int nc = lane & 15;
    v8f cc[4] = {c0, c1, c2, c3};
#pragma unroll
    for (int j = 0; j < 4; ++j) {
#pragma unroll
        for (int r = 0; r < 8; ++r) {
            int mrow = m0 + hi + r;
            int n    = n0 + j * 16 + nc;
            int bb = mrow >> 12, s = mrow & (SEQ - 1);
            int hh = n >> 6,     d = n & 63;
            float v = cc[j][r];
            size_t bhs = (size_t)(bb * NH + hh) * SEQ + s;
            if (z == 0)      Qp[bhs * DK + d] = f2bf(v * 0.125f);
            else if (z == 1) Kp[bhs * DK + d] = f2bf(v);
            else Vtp[((size_t)(bb * NH + hh) * DK + d) * SEQ + s] = f2bf(v);
        }
    }
}

// ---------------------------------------------------------------------------
// Flash attention with async-staged, double-buffered K/V tiles in LDS.
// Block = 4 waves, same (b,h); each wave owns a 16-row query tile.
// Per 32-key step: 4 score WMMAs, online softmax, P->LDS->A-frag, 4 PV WMMAs.
// ---------------------------------------------------------------------------
__device__ __forceinline__ void stage_async(const unsigned short* __restrict__ Kh,
                                            const unsigned short* __restrict__ Vh,
                                            int k0, int tid,
                                            unsigned short* kd, unsigned short* vd) {
    // K tile: 32 rows x 64 halves = 256 chunks of 16B; thread -> chunks tid, tid+128
    async_ld16(kd + (size_t)tid * 8,
               Kh + (size_t)(k0 + (tid >> 3)) * DK + ((tid & 7) << 3));
    async_ld16(kd + (size_t)(tid + 128) * 8,
               Kh + (size_t)(k0 + ((tid + 128) >> 3)) * DK + (((tid + 128) & 7) << 3));
    // V^T tile: 64 rows x 32 halves = 256 chunks of 16B
    async_ld16(vd + (size_t)tid * 8,
               Vh + (size_t)(tid >> 2) * SEQ + k0 + ((tid & 3) << 3));
    async_ld16(vd + (size_t)(tid + 128) * 8,
               Vh + (size_t)((tid + 128) >> 2) * SEQ + k0 + (((tid + 128) & 3) << 3));
}

__global__ void __launch_bounds__(128)
mha_flash_attn(const unsigned short* __restrict__ Qp,
               const unsigned short* __restrict__ Kp,
               const unsigned short* __restrict__ Vtp,
               unsigned short* __restrict__ Ob) {
    __shared__ unsigned short kbuf[2][32 * 64];   // 4KB per buffer
    __shared__ unsigned short vbuf[2][64 * 32];   // 4KB per buffer
    __shared__ unsigned short pbuf[4 * 16 * 32];  // per-wave P tiles
    const int tid  = threadIdx.x;
    const int wave = tid >> 5;
    const int lane = tid & 31;
    const int h = blockIdx.y, b = blockIdx.z;
    const int bh = b * NH + h;
    const int q0 = (blockIdx.x * 4 + wave) * 16;
    const unsigned short* Qh = Qp + (size_t)bh * SEQ * DK;
    const unsigned short* Kh = Kp + (size_t)bh * SEQ * DK;
    const unsigned short* Vh = Vtp + (size_t)bh * DK * SEQ;
    unsigned short* myp = pbuf + wave * 512;

    // Kick off tile 0 immediately, then load Q while it is in flight.
    stage_async(Kh, Vh, 0, tid, kbuf[0], vbuf[0]);

    v16bf qa0 = load_frag(Qh, q0, DK, 0);    // Q rows, d 0..31 (pre-scaled)
    v16bf qa1 = load_frag(Qh, q0, DK, 32);   // Q rows, d 32..63
    v8f o0 = {}, o1 = {}, o2 = {}, o3 = {};
    float m[8], l[8];
#pragma unroll
    for (int r = 0; r < 8; ++r) { m[r] = -3.0e38f; l[r] = 0.0f; }
    const int hi = (lane >> 4) << 3;
    const int nc = lane & 15;

    wait_async0();
    __syncthreads();

    const int NSTEP = SEQ / 32;
    for (int i = 0; i < NSTEP; ++i) {
        const int cur = i & 1;
        const unsigned short* kb = kbuf[cur];
        const unsigned short* vb = vbuf[cur];
        if (i + 1 < NSTEP)                              // tile i+1 in flight
            stage_async(Kh, Vh, (i + 1) * 32, tid, kbuf[cur ^ 1], vbuf[cur ^ 1]);

        // ---- scores: 16 queries x 32 keys, accumulated over d=64 ----
        v8f s0 = {}, s1 = {};
        s0 = wmma_bf16(qa0, load_frag(kb, 0,  DK, 0),  s0);
        s0 = wmma_bf16(qa1, load_frag(kb, 0,  DK, 32), s0);
        s1 = wmma_bf16(qa0, load_frag(kb, 16, DK, 0),  s1);
        s1 = wmma_bf16(qa1, load_frag(kb, 16, DK, 32), s1);

        // ---- online softmax (row stats live per 16-lane half) ----
#pragma unroll
        for (int r = 0; r < 8; ++r) {
            float t = fmaxf(s0[r], s1[r]);
            t = fmaxf(t, __shfl_xor(t, 1));
            t = fmaxf(t, __shfl_xor(t, 2));
            t = fmaxf(t, __shfl_xor(t, 4));
            t = fmaxf(t, __shfl_xor(t, 8));
            float nm = fmaxf(m[r], t);
            float al = __expf(m[r] - nm);
            m[r] = nm;
            float p0 = __expf(s0[r] - nm);
            float p1 = __expf(s1[r] - nm);
            s0[r] = p0; s1[r] = p1;
            float cs = p0 + p1;
            cs += __shfl_xor(cs, 1);
            cs += __shfl_xor(cs, 2);
            cs += __shfl_xor(cs, 4);
            cs += __shfl_xor(cs, 8);
            l[r] = l[r] * al + cs;
            o0[r] *= al; o1[r] *= al; o2[r] *= al; o3[r] *= al;
        }

        // ---- P (C-layout) -> wave-private LDS, reload as one A-fragment ----
#pragma unroll
        for (int r = 0; r < 8; ++r) {
            myp[(hi + r) * 32 + nc]      = f2bf(s0[r]);
            myp[(hi + r) * 32 + nc + 16] = f2bf(s1[r]);
        }
        asm volatile("s_wait_dscnt 0" ::: "memory");
        v16bf pa = load_frag(myp, 0, 32, 0);

        // ---- O += P x V ----
        o0 = wmma_bf16(pa, load_frag(vb, 0,  32, 0), o0);
        o1 = wmma_bf16(pa, load_frag(vb, 16, 32, 0), o1);
        o2 = wmma_bf16(pa, load_frag(vb, 32, 32, 0), o2);
        o3 = wmma_bf16(pa, load_frag(vb, 48, 32, 0), o3);

        if (i + 1 < NSTEP) wait_async0();   // tile i+1 landed
        __syncthreads();                    // visible block-wide before use
    }

    // ---- normalize and store merged heads [B,S,768] as bf16 ----
    v8f oo[4] = {o0, o1, o2, o3};
#pragma unroll
    for (int r = 0; r < 8; ++r) {
        float rl = 1.0f / l[r];
        size_t row = (size_t)b * SEQ + q0 + hi + r;
#pragma unroll
        for (int j = 0; j < 4; ++j) {
            Ob[row * D_EMB + h * DK + j * 16 + nc] = f2bf(oo[j][r] * rl);
        }
    }
}

// ---------------------------------------------------------------------------
// Output projection, software-pipelined: O_bf16 [8192x768] x Wo -> fp32 out.
// ---------------------------------------------------------------------------
__global__ void __launch_bounds__(128)
mha_out_proj(const unsigned short* __restrict__ Ob,
             const unsigned short* __restrict__ Wot,
             float* __restrict__ out) {
    const int wave = threadIdx.x >> 5;
    const int lane = threadIdx.x & 31;
    const int m0 = blockIdx.x * 64 + wave * 16;
    const int n0 = blockIdx.y * 64;

    v8f c0 = {}, c1 = {}, c2 = {}, c3 = {};
    v16bf a  = load_frag(Ob, m0, D_EMB, 0);
    v16bf b0 = load_frag(Wot, n0,      D_EMB, 0);
    v16bf b1 = load_frag(Wot, n0 + 16, D_EMB, 0);
    v16bf b2 = load_frag(Wot, n0 + 32, D_EMB, 0);
    v16bf b3 = load_frag(Wot, n0 + 48, D_EMB, 0);
    for (int k = 32; k < D_EMB; k += 32) {
        v16bf an  = load_frag(Ob, m0, D_EMB, k);
        v16bf bn0 = load_frag(Wot, n0,      D_EMB, k);
        v16bf bn1 = load_frag(Wot, n0 + 16, D_EMB, k);
        v16bf bn2 = load_frag(Wot, n0 + 32, D_EMB, k);
        v16bf bn3 = load_frag(Wot, n0 + 48, D_EMB, k);
        c0 = wmma_bf16(a, b0, c0);
        c1 = wmma_bf16(a, b1, c1);
        c2 = wmma_bf16(a, b2, c2);
        c3 = wmma_bf16(a, b3, c3);
        a = an; b0 = bn0; b1 = bn1; b2 = bn2; b3 = bn3;
    }
    c0 = wmma_bf16(a, b0, c0);
    c1 = wmma_bf16(a, b1, c1);
    c2 = wmma_bf16(a, b2, c2);
    c3 = wmma_bf16(a, b3, c3);

    const int hi = (lane >> 4) << 3;
    const int nc = lane & 15;
    v8f cc[4] = {c0, c1, c2, c3};
#pragma unroll
    for (int j = 0; j < 4; ++j)
#pragma unroll
        for (int r = 0; r < 8; ++r)
            out[(size_t)(m0 + hi + r) * D_EMB + n0 + j * 16 + nc] = cc[j][r];
}

// ---------------------------------------------------------------------------
extern "C" void kernel_launch(void* const* d_in, const int* in_sizes, int n_in,
                              void* d_out, int out_size, void* d_ws, size_t ws_size,
                              hipStream_t stream) {
    const float* X  = (const float*)d_in[0];
    const float* Wq = (const float*)d_in[1];
    const float* Wk = (const float*)d_in[2];
    const float* Wv = (const float*)d_in[3];
    const float* Wo = (const float*)d_in[4];
    float* out = (float*)d_out;

    const int    M  = 2 * SEQ;                 // 8192 rows
    const size_t nX = (size_t)M * D_EMB;       // 6 291 456
    const size_t nW = (size_t)D_EMB * D_EMB;   //   589 824

    // workspace layout (bf16 halves): X | 4x W^T | Q | K | V^T | O  (~68 MB)
    unsigned short* Xb  = (unsigned short*)d_ws;
    unsigned short* Wt  = Xb + nX;
    unsigned short* Qp  = Wt + 4 * nW;
    unsigned short* Kp  = Qp + nX;
    unsigned short* Vtp = Kp + nX;
    unsigned short* Ob  = Vtp + nX;

    mha_cvt_f32_bf16<<<(int)(nX / 256), 256, 0, stream>>>(X, Xb, (int)nX);
    mha_transpose_w<<<(int)(nW / 256), 256, 0, stream>>>(Wq, Wt + 0 * nW);
    mha_transpose_w<<<(int)(nW / 256), 256, 0, stream>>>(Wk, Wt + 1 * nW);
    mha_transpose_w<<<(int)(nW / 256), 256, 0, stream>>>(Wv, Wt + 2 * nW);
    mha_transpose_w<<<(int)(nW / 256), 256, 0, stream>>>(Wo, Wt + 3 * nW);

    mha_qkv_gemm<<<dim3(M / 64, D_EMB / 64, 3), 128, 0, stream>>>(Xb, Wt, Qp, Kp, Vtp);
    mha_flash_attn<<<dim3(SEQ / 64, NH, 2), 128, 0, stream>>>(Qp, Kp, Vtp, Ob);
    mha_out_proj<<<dim3(M / 64, D_EMB / 64), 128, 0, stream>>>(Ob, Wt + 3 * nW, out);
}
